// EConditionalGraphDenoisingNetwork_75857712382447
// MI455X (gfx1250) — compile-verified, matrix operand order checked by reading, not run on hardware
//
#include <hip/hip_runtime.h>
#include <hip/hip_bf16.h>
#include <math.h>
#include <stdint.h>

// ---------------- problem constants ----------------
#define N_NODES   100
#define N_EDGES   100000
#define HID       256
#define DIN       128          // NODE_FEAT*PRED_H
#define CIN       6            // POS_DIM*OBS_H
#define NLAYERS   5
#define EPSV      1e-8f
#define PRED_OUT  128

typedef __bf16 bf16;
typedef __attribute__((ext_vector_type(16))) __bf16 v16bf;
typedef __attribute__((ext_vector_type(8)))  __bf16 v8bf;
typedef __attribute__((ext_vector_type(8)))  float  v8f;

// ---------------- workspace layout (floats) ----------------
#define WS_H       0                 // h:    100*256
#define WS_A       25600             // A:    100*256
#define WS_B       51200             // B:    100*256
#define WS_MAGG    76800             // magg: 100*256
#define WS_SCALE   102400            // 256
#define WS_BIAS    102656            // 256
#define WS_WR      102912            // 256
#define WS_WE      103168            // 256
#define WS_C0      103424            // 256
#define WS_CW2     103680            // 256
#define WS_AGG     103936            // 300
#define WS_CNT     104236            // 100
#define WS_COORD   104336            // 300
#define WS_Z       104640            // 100*256
#define WS_G       130240            // 256
#define BF_BYTE_OFF (560*1024)       // bf16 region: eW2t (65536) then cW1t (65536)

static __device__ __forceinline__ float siluf(float x) {
    // x * rcp(1+exp(-x)) : v_exp_f32 + v_rcp_f32, no IEEE divide expansion
    return x * __builtin_amdgcn_rcpf(1.0f + __expf(-x));
}
static __device__ __forceinline__ float mishf(float x) {
    float sp = log1pf(__expf(x));
    return x * tanhf(sp);
}
static __device__ __forceinline__ v16bf mk16(v8bf lo, v8bf hi) {
    v16bf r;
#pragma unroll
    for (int i = 0; i < 8; ++i) { r[i] = lo[i]; r[i + 8] = hi[i]; }
    return r;
}

// Issue one 8KB weight panel (16 cols x 256 K, bf16, contiguous) into LDS
// via CDNA5 async copy.  128 threads x 4 chunks x 16B = 8192B.  Each wave
// issues 4 async instructions -> ASYNCcnt += 4 per wave.
static __device__ __forceinline__ void issue_panel(const bf16* __restrict__ Wt,
                                                   int nt, bf16* panel, int tid) {
    const char* gp = (const char*)(Wt + nt * 16 * HID);
    unsigned lbase = (unsigned)(uintptr_t)panel;   // low 32 bits = LDS offset
#pragma unroll
    for (int i = 0; i < 4; ++i) {
        unsigned off = (unsigned)(tid + i * 128) * 16u;
        unsigned laddr = lbase + off;
        unsigned long long ga = (unsigned long long)(uintptr_t)(gp + off);
        asm volatile("global_load_async_to_lds_b128 %0, %1, off"
                     :: "v"(laddr), "v"(ga) : "memory");
    }
}
static __device__ __forceinline__ void wait_async_le4() {
    asm volatile("s_wait_asynccnt 0x4" ::: "memory");
}
static __device__ __forceinline__ void wait_async_0() {
    asm volatile("s_wait_asynccnt 0x0" ::: "memory");
}

// ================= node embedding + positional encoding =================
__global__ void k_node_emb(const float* __restrict__ x,
                           const float* __restrict__ W,
                           const float* __restrict__ b,
                           float* __restrict__ h) {
    int i = blockIdx.x;          // node
    int j = threadIdx.x;         // hidden col
    float acc = b[j];
    const float* xr = x + i * DIN;
#pragma unroll 4
    for (int k = 0; k < DIN; ++k) acc += xr[k] * W[k * HID + j];
    // pe[N_NODES-1][j], pos = 100
    float d = __expf(-(float)(2 * (j >> 1)) * (logf(10000.0f) / (float)HID));
    float ang = 100.0f * d;
    acc += (j & 1) ? cosf(ang) : sinf(ang);
    h[i * HID + j] = acc;
}

// ================= init working coord copy =================
__global__ void k_coord_init(const float* __restrict__ xc, float* __restrict__ coord) {
    int t = blockIdx.x * blockDim.x + threadIdx.x;
    if (t < N_NODES * 3) coord[t] = xc[t];
}

// ================= FiLM conditioning (one block, 512 threads) =================
__global__ void k_film(const float* __restrict__ cond,
                       const float* __restrict__ cW,  // 6 x 512
                       const float* __restrict__ cb,  // 512
                       float* __restrict__ scale, float* __restrict__ bias) {
    __shared__ float msum[CIN];
    int t = threadIdx.x;
    if (t < CIN) {
        float s = 0.f;
        for (int i = 0; i < N_NODES; ++i) s += mishf(cond[i * CIN + t]);
        msum[t] = s;
    }
    __syncthreads();
    float acc = 0.f;
#pragma unroll
    for (int k = 0; k < CIN; ++k) acc += msum[k] * cW[k * (2 * HID) + t];
    float e = cb[t] + acc * (1.0f / (float)N_NODES);
    if (t < HID) scale[t] = e; else bias[t - HID] = e;
}

// ================= h = scale*h + bias =================
__global__ void k_scale(float* __restrict__ h, const float* __restrict__ scale,
                        const float* __restrict__ bias) {
    int i = blockIdx.x, j = threadIdx.x;
    h[i * HID + j] = scale[j] * h[i * HID + j] + bias[j];
}

// ================= node tables A = h@eW1[0:256], B = h@eW1[256:512] =================
__global__ void k_ab(const float* __restrict__ h, const float* __restrict__ eW1,
                     float* __restrict__ A, float* __restrict__ B) {
    int i = blockIdx.x, j = threadIdx.x;
    const float* hr = h + i * HID;
    float a = 0.f, bb = 0.f;
#pragma unroll 4
    for (int k = 0; k < HID; ++k) {
        float hv = hr[k];
        a  += hv * eW1[k * HID + j];
        bb += hv * eW1[(HID + k) * HID + j];
    }
    A[i * HID + j] = a;
    B[i * HID + j] = bb;
}

// ================= rank-1 vectors: w_r, w_e, c0, cW2 copy =================
__global__ void k_vecs(const float* __restrict__ eW1, const float* __restrict__ eb1,
                       const float* __restrict__ eW,  const float* __restrict__ eb,
                       const float* __restrict__ cW2,
                       float* __restrict__ wr, float* __restrict__ we,
                       float* __restrict__ c0, float* __restrict__ cw2v) {
    int j = threadIdx.x;
    wr[j] = eW1[512 * HID + j];
    float a = 0.f, c = eb1[j];
#pragma unroll 4
    for (int k = 0; k < HID; ++k) {
        float w3 = eW1[(513 + k) * HID + j];
        a += eW[k] * w3;
        c += eb[k] * w3;
    }
    we[j] = a;
    c0[j] = c;
    cw2v[j] = cW2[j];
}

// ================= weight transpose+convert to bf16 (N-major for WMMA B) ==========
__global__ void k_wcvt(const float* __restrict__ eW2, const float* __restrict__ cW1,
                       bf16* __restrict__ eW2t, bf16* __restrict__ cW1t) {
    int t = blockIdx.x * blockDim.x + threadIdx.x;   // t = n*256 + k
    int n = t >> 8, k = t & 255;
    eW2t[t] = (bf16)eW2[k * HID + n];
    cW1t[t] = (bf16)cW1[k * HID + n];
}

// ================= zero accumulators =================
__global__ void k_zero(float* __restrict__ magg, float* __restrict__ agg,
                       float* __restrict__ cnt) {
    int t = blockIdx.x * blockDim.x + threadIdx.x;
    if (t < N_NODES * HID) magg[t] = 0.f;
    if (t < N_NODES * 3)   agg[t]  = 0.f;
    if (t < N_NODES)       cnt[t]  = 0.f;
}

// ====== fused edge kernel: 128 edges/block, 32 edges/wave (2 M-tiles/wave) ======
__global__ void __launch_bounds__(128)
k_edge(const int*  __restrict__ ei,      // 2 x E
       const float* __restrict__ ea,     // E
       const float* __restrict__ coord,  // N x 3
       const float* __restrict__ A, const float* __restrict__ B,
       const float* __restrict__ wr, const float* __restrict__ we,
       const float* __restrict__ c0,
       const float* __restrict__ eb2, const float* __restrict__ cb1,
       const float* __restrict__ cw2v,
       const bf16* __restrict__ eW2t, const bf16* __restrict__ cW1t,
       float* __restrict__ magg, float* __restrict__ agg, float* __restrict__ cnt) {
    __shared__ __align__(16) bf16 sPre[4][32][HID];      // 64 KB
    __shared__ __align__(16) bf16 sM[4][32][HID];        // 64 KB
    __shared__ __align__(16) bf16 sPanel[2][16 * HID];   // 16 KB double-buffered
    __shared__ float sCd[4][32][3];
    __shared__ int   sRow[4][32];
    __shared__ int   sVal[4][32];

    const int tid  = threadIdx.x;
    const int w    = tid >> 5;
    const int lane = tid & 31;
    const int e0   = blockIdx.x * 128 + w * 32;

    // ---------- stage 1: build silu'd input tile (32 x 256), one edge per lane ----
    {
        int e = e0 + lane;
        bool valid = (e < N_EDGES);
        int ecl = valid ? e : 0;
        int r = ei[ecl];
        int c = ei[N_EDGES + ecl];
        float eav = valid ? ea[ecl] : 0.f;
        float dx = coord[r * 3 + 0] - coord[c * 3 + 0];
        float dy = coord[r * 3 + 1] - coord[c * 3 + 1];
        float dz = coord[r * 3 + 2] - coord[c * 3 + 2];
        float radial = dx * dx + dy * dy + dz * dz;
        float inv = 1.0f / (sqrtf(radial) + EPSV);
        sRow[w][lane] = r;
        sVal[w][lane] = valid ? 1 : 0;
        sCd[w][lane][0] = dx * inv;
        sCd[w][lane][1] = dy * inv;
        sCd[w][lane][2] = dz * inv;
        const float4* Ar = (const float4*)(A + r * HID);
        const float4* Bc = (const float4*)(B + c * HID);
        const float4* Wr = (const float4*)wr;
        const float4* We = (const float4*)we;
        const float4* C0 = (const float4*)c0;
#pragma unroll 4
        for (int j4 = 0; j4 < 64; ++j4) {
            float4 av = Ar[j4], bv = Bc[j4], wv = Wr[j4], ev = We[j4], cv = C0[j4];
            float v0 = av.x + bv.x + radial * wv.x + eav * ev.x + cv.x;
            float v1 = av.y + bv.y + radial * wv.y + eav * ev.y + cv.y;
            float v2 = av.z + bv.z + radial * wv.z + eav * ev.z + cv.z;
            float v3 = av.w + bv.w + radial * wv.w + eav * ev.w + cv.w;
            int col = j4 * 4;
            sPre[w][lane][col + 0] = (bf16)(valid ? siluf(v0) : 0.f);
            sPre[w][lane][col + 1] = (bf16)(valid ? siluf(v1) : 0.f);
            sPre[w][lane][col + 2] = (bf16)(valid ? siluf(v2) : 0.f);
            sPre[w][lane][col + 3] = (bf16)(valid ? siluf(v3) : 0.f);
        }
    }
    __syncthreads();

    const int nrow = lane & 15;   // C/D column within 16-tile; also A/B lane row
    const int hi   = lane >> 4;

    // hoist per-edge row/valid for this lane's rows (2 M-tiles x 8 rows)
    int rr[2][8];
    int vv[2][8];
#pragma unroll
    for (int mt = 0; mt < 2; ++mt) {
#pragma unroll
        for (int i = 0; i < 8; ++i) {
            int mr = mt * 16 + hi * 8 + i;
            rr[mt][i] = sRow[w][mr];
            vv[mt][i] = sVal[w][mr];
        }
    }

    // ---------- stage 2: m = silu(pre @ eW2 + eb2), magg scatter ----------
    issue_panel(eW2t, 0, &sPanel[0][0], tid);
    for (int nt = 0; nt < 16; ++nt) {
        if (nt < 15) { issue_panel(eW2t, nt + 1, &sPanel[(nt + 1) & 1][0], tid); wait_async_le4(); }
        else         { wait_async_0(); }
        __syncthreads();                       // panel nt visible to all waves
        const bf16* Bp = &sPanel[nt & 1][0];
        int ncol = nt * 16 + nrow;
        float bvs = eb2[ncol];
        v8f acc0, acc1;
#pragma unroll
        for (int i = 0; i < 8; ++i) { acc0[i] = bvs; acc1[i] = bvs; }
#pragma unroll
        for (int kt = 0; kt < 8; ++kt) {
            const bf16* bp = Bp + nrow * HID + kt * 32 + hi * 8;
            v16bf b = mk16(*(const v8bf*)bp, *(const v8bf*)(bp + 16));
            const bf16* a0p = &sPre[w][nrow][kt * 32 + hi * 8];
            v16bf a0 = mk16(*(const v8bf*)a0p, *(const v8bf*)(a0p + 16));
            acc0 = __builtin_amdgcn_wmma_f32_16x16x32_bf16(
                false, a0, false, b, (short)0, acc0, false, false);
            const bf16* a1p = &sPre[w][16 + nrow][kt * 32 + hi * 8];
            v16bf a1 = mk16(*(const v8bf*)a1p, *(const v8bf*)(a1p + 16));
            acc1 = __builtin_amdgcn_wmma_f32_16x16x32_bf16(
                false, a1, false, b, (short)0, acc1, false, false);
        }
#pragma unroll
        for (int i = 0; i < 8; ++i) {
            int mr = hi * 8 + i;
            float s0 = siluf(acc0[i]);
            float s1 = siluf(acc1[i]);
            sM[w][mr][ncol] = (bf16)s0;
            sM[w][16 + mr][ncol] = (bf16)s1;
            if (vv[0][i]) atomicAdd(&magg[rr[0][i] * HID + ncol], s0);
            if (vv[1][i]) atomicAdd(&magg[rr[1][i] * HID + ncol], s1);
        }
        __syncthreads();                       // done reading panel before reuse
    }

    // ---------- stage 3: cm = silu(m @ cW1 + cb1); s = cm @ cW2; coord scatter ----
    float part[2][8];
#pragma unroll
    for (int mt = 0; mt < 2; ++mt)
#pragma unroll
        for (int i = 0; i < 8; ++i) part[mt][i] = 0.f;

    issue_panel(cW1t, 0, &sPanel[0][0], tid);
    for (int nt = 0; nt < 16; ++nt) {
        if (nt < 15) { issue_panel(cW1t, nt + 1, &sPanel[(nt + 1) & 1][0], tid); wait_async_le4(); }
        else         { wait_async_0(); }
        __syncthreads();
        const bf16* Bp = &sPanel[nt & 1][0];
        int ncol = nt * 16 + nrow;
        float bvs = cb1[ncol];
        v8f acc0, acc1;
#pragma unroll
        for (int i = 0; i < 8; ++i) { acc0[i] = bvs; acc1[i] = bvs; }
#pragma unroll
        for (int kt = 0; kt < 8; ++kt) {
            const bf16* bp = Bp + nrow * HID + kt * 32 + hi * 8;
            v16bf b = mk16(*(const v8bf*)bp, *(const v8bf*)(bp + 16));
            const bf16* a0p = &sM[w][nrow][kt * 32 + hi * 8];
            v16bf a0 = mk16(*(const v8bf*)a0p, *(const v8bf*)(a0p + 16));
            acc0 = __builtin_amdgcn_wmma_f32_16x16x32_bf16(
                false, a0, false, b, (short)0, acc0, false, false);
            const bf16* a1p = &sM[w][16 + nrow][kt * 32 + hi * 8];
            v16bf a1 = mk16(*(const v8bf*)a1p, *(const v8bf*)(a1p + 16));
            acc1 = __builtin_amdgcn_wmma_f32_16x16x32_bf16(
                false, a1, false, b, (short)0, acc1, false, false);
        }
        float cw = cw2v[ncol];
#pragma unroll
        for (int i = 0; i < 8; ++i) {
            part[0][i] += siluf(acc0[i]) * cw;
            part[1][i] += siluf(acc1[i]) * cw;
        }
        __syncthreads();
    }
    // reduce over the 16 lanes (columns) within each half-wave
#pragma unroll
    for (int off = 8; off >= 1; off >>= 1) {
#pragma unroll
        for (int i = 0; i < 8; ++i) {
            part[0][i] += __shfl_xor(part[0][i], off, 32);
            part[1][i] += __shfl_xor(part[1][i], off, 32);
        }
    }
    if ((lane & 15) < 8) {
        int i = lane & 7;
#pragma unroll
        for (int mt = 0; mt < 2; ++mt) {
            int mr = mt * 16 + hi * 8 + i;
            if (sVal[w][mr]) {
                int r = sRow[w][mr];
                float sv = part[mt][i];
                atomicAdd(&agg[r * 3 + 0], sCd[w][mr][0] * sv);
                atomicAdd(&agg[r * 3 + 1], sCd[w][mr][1] * sv);
                atomicAdd(&agg[r * 3 + 2], sCd[w][mr][2] * sv);
                atomicAdd(&cnt[r], 1.0f);
            }
        }
    }
}

// ================= coord += agg / max(cnt,1) =================
__global__ void k_coord_upd(float* __restrict__ coord, const float* __restrict__ agg,
                            const float* __restrict__ cnt) {
    int i = threadIdx.x;
    if (i < N_NODES) {
        float c = fmaxf(cnt[i], 1.0f);
        coord[i * 3 + 0] += agg[i * 3 + 0] / c;
        coord[i * 3 + 1] += agg[i * 3 + 1] / c;
        coord[i * 3 + 2] += agg[i * 3 + 2] / c;
    }
}

// ================= z = silu([h,magg] @ nW1 + nb1) =================
__global__ void k_zmlp(const float* __restrict__ h, const float* __restrict__ magg,
                       const float* __restrict__ nW1, const float* __restrict__ nb1,
                       float* __restrict__ z) {
    int i = blockIdx.x, j = threadIdx.x;
    float acc = nb1[j];
    const float* hr = h + i * HID;
    const float* mr = magg + i * HID;
#pragma unroll 4
    for (int k = 0; k < HID; ++k) {
        acc += hr[k] * nW1[k * HID + j];
        acc += mr[k] * nW1[(HID + k) * HID + j];
    }
    z[i * HID + j] = siluf(acc);
}

// ================= h += z @ nW2 + nb2 =================
__global__ void k_hupd(float* __restrict__ h, const float* __restrict__ z,
                       const float* __restrict__ nW2, const float* __restrict__ nb2) {
    int i = blockIdx.x, j = threadIdx.x;
    float acc = nb2[j];
    const float* zr = z + i * HID;
#pragma unroll 4
    for (int k = 0; k < HID; ++k) acc += zr[k] * nW2[k * HID + j];
    h[i * HID + j] += acc;
}

// ================= g = mean(h, axis=0) =================
__global__ void k_gmean(const float* __restrict__ h, float* __restrict__ g) {
    int j = threadIdx.x;
    float s = 0.f;
    for (int i = 0; i < N_NODES; ++i) s += h[i * HID + j];
    g[j] = s * (1.0f / (float)N_NODES);
}

// ================= prediction head (row N-1 only) =================
__global__ void k_pred(const float* __restrict__ g, const float* __restrict__ h,
                       const float* __restrict__ W1, const float* __restrict__ b1,
                       const float* __restrict__ W2, const float* __restrict__ b2,
                       const float* __restrict__ W3, const float* __restrict__ b3,
                       float* __restrict__ out) {
    __shared__ float in512[2 * HID];
    __shared__ float z1[HID];
    __shared__ float z2[HID];
    int j = threadIdx.x;
    in512[j] = g[j];
    in512[HID + j] = h[(N_NODES - 1) * HID + j];
    __syncthreads();
    float a = b1[j];
#pragma unroll 4
    for (int k = 0; k < 2 * HID; ++k) a += in512[k] * W1[k * HID + j];
    z1[j] = fmaxf(a, 0.f);
    __syncthreads();
    a = b2[j];
#pragma unroll 4
    for (int k = 0; k < HID; ++k) a += z1[k] * W2[k * HID + j];
    z2[j] = fmaxf(a, 0.f);
    __syncthreads();
    if (j < PRED_OUT) {
        a = b3[j];
#pragma unroll 4
        for (int k = 0; k < HID; ++k) a += z2[k] * W3[k * PRED_OUT + j];
        out[j] = a;
    }
}

__global__ void k_outcoord(const float* __restrict__ coord, float* __restrict__ out) {
    int t = blockIdx.x * blockDim.x + threadIdx.x;
    if (t < N_NODES * 3) out[PRED_OUT + t] = coord[t];
}

// ======================================================================
extern "C" void kernel_launch(void* const* d_in, const int* in_sizes, int n_in,
                              void* d_out, int out_size, void* d_ws, size_t ws_size,
                              hipStream_t stream) {
    (void)in_sizes; (void)n_in; (void)out_size; (void)ws_size;
    const float* x     = (const float*)d_in[0];
    const int*   ei    = (const int*)  d_in[1];
    const float* ea    = (const float*)d_in[2];
    const float* xc    = (const float*)d_in[3];
    const float* cond  = (const float*)d_in[4];
    // params (jax pytree, sorted dict keys): cond, edge_emb, layers, node_emb, pred
    const int PB = 5;
    auto P = [&](int i) { return (const float*)d_in[i]; };
    const float* eeW = P(PB + 10);   // edge_emb W (1x256)
    const float* eeb = P(PB + 11);   // edge_emb b (256)
    const float* neW = P(PB + 67);   // node_emb W (128x256)
    const float* neb = P(PB + 68);   // node_emb b
    const float* pW1 = P(PB + 69); const float* pb1 = P(PB + 70);
    const float* pW2 = P(PB + 71); const float* pb2 = P(PB + 72);
    const float* pW3 = P(PB + 73); const float* pb3 = P(PB + 74);

    float* F = (float*)d_ws;
    float* h     = F + WS_H;
    float* A     = F + WS_A;
    float* B     = F + WS_B;
    float* magg  = F + WS_MAGG;
    float* scale = F + WS_SCALE;
    float* bias  = F + WS_BIAS;
    float* wr    = F + WS_WR;
    float* we    = F + WS_WE;
    float* c0    = F + WS_C0;
    float* cw2v  = F + WS_CW2;
    float* agg   = F + WS_AGG;
    float* cnt   = F + WS_CNT;
    float* coord = F + WS_COORD;
    float* z     = F + WS_Z;
    float* g     = F + WS_G;
    bf16* eW2t = (bf16*)((char*)d_ws + BF_BYTE_OFF);
    bf16* cW1t = eW2t + HID * HID;
    float* out = (float*)d_out;

    k_node_emb<<<N_NODES, HID, 0, stream>>>(x, neW, neb, h);
    k_coord_init<<<2, 256, 0, stream>>>(xc, coord);

    for (int l = 0; l < NLAYERS; ++l) {
        int lb = PB + 12 + 11 * l;
        const float* cW1 = P(lb + 0);
        const float* cW2 = P(lb + 1);
        const float* cb1 = P(lb + 2);
        const float* eW1 = P(lb + 3);
        const float* eW2 = P(lb + 4);
        const float* eb1 = P(lb + 5);
        const float* eb2 = P(lb + 6);
        const float* nW1 = P(lb + 7);
        const float* nW2 = P(lb + 8);
        const float* nb1 = P(lb + 9);
        const float* nb2 = P(lb + 10);
        const float* ccW = P(PB + 2 * l);
        const float* ccb = P(PB + 2 * l + 1);

        k_film<<<1, 2 * HID, 0, stream>>>(cond, ccW, ccb, scale, bias);
        k_scale<<<N_NODES, HID, 0, stream>>>(h, scale, bias);
        k_ab<<<N_NODES, HID, 0, stream>>>(h, eW1, A, B);
        k_vecs<<<1, HID, 0, stream>>>(eW1, eb1, eeW, eeb, cW2, wr, we, c0, cw2v);
        k_wcvt<<<(HID * HID) / 256, 256, 0, stream>>>(eW2, cW1, eW2t, cW1t);
        k_zero<<<(N_NODES * HID + 255) / 256, 256, 0, stream>>>(magg, agg, cnt);
        k_edge<<<(N_EDGES + 127) / 128, 128, 0, stream>>>(
            ei, ea, coord, A, B, wr, we, c0, eb2, cb1, cw2v,
            eW2t, cW1t, magg, agg, cnt);
        k_coord_upd<<<1, 128, 0, stream>>>(coord, agg, cnt);
        k_zmlp<<<N_NODES, HID, 0, stream>>>(h, magg, nW1, nb1, z);
        k_hupd<<<N_NODES, HID, 0, stream>>>(h, z, nW2, nb2);
    }

    k_gmean<<<1, HID, 0, stream>>>(h, g);
    k_pred<<<1, HID, 0, stream>>>(g, h, pW1, pb1, pW2, pb2, pW3, pb3, out);
    k_outcoord<<<2, 256, 0, stream>>>(coord, out);
}
